// BothMamba_17325898072384
// MI455X (gfx1250) — compile-verified, matrix operand order checked by reading, not run on hardware
//
#include <hip/hip_runtime.h>
#include <cstdint>
#include <cstddef>

// ---------------------------------------------------------------------------
// BothMamba (spatial row+col Mamba -> GN+SiLU) fused with spectral Mamba,
// softmax-weighted sum.  gfx1250 / CDNA5: wave32, v_wmma_f32_16x16x32_f16 for
// all dense projections, LDS staging via async global->LDS DMA when available.
//
// Input flattening assumption (setup_inputs dict order, nested dicts in
// insertion order):
//   0: x (8,96,128,128)
//   1..9:   row_p  {in_proj_w, conv_w, conv_b, x_proj_w, dt_proj_w,
//                   dt_proj_b, A_log, D, out_proj_w}
//   10..18: col_p  (same 9)
//   19..27: spe_p  (same 9)
//   28: spa_gn_g  29: spa_gn_b  30: spe_gn_g  31: spe_gn_b
//   32: pe (1,12) 33: fuse_w (2)
// Output: f32 (8,96,128,128)
// ---------------------------------------------------------------------------

typedef __attribute__((ext_vector_type(16))) _Float16 v16h;
typedef __attribute__((ext_vector_type(8)))  _Float16 v8h;
typedef __attribute__((ext_vector_type(8)))  float    v8f;
typedef int vi4 __attribute__((__vector_size__(16)));   // 16B chunk for LDS DMA

static constexpr int Bn = 8, Cn = 96, Hn = 128, Wn = 128;
static constexpr int TOKN = 8, GDIM = 12;        // spectral tokens / group dim
static constexpr float GN_EPS = 1e-5f;

#if defined(__has_builtin)
# if __has_builtin(__builtin_amdgcn_global_load_async_to_lds_b128) && \
     __has_builtin(__builtin_amdgcn_s_wait_asynccnt)
#  define ATHENA_ASYNC_LDS 1
# endif
#endif

__device__ __forceinline__ float siluf(float v) {
  return v * (1.0f / (1.0f + __expf(-v)));
}

// Load one WMMA f16 operand fragment from an LDS row pointer.
// lane<16 -> K chunks {k0..k0+7, k0+16..k0+23}; lane>=16 shifts by 8 (sel).
__device__ __forceinline__ v16h ld_frag(const _Float16* row, int k0) {
  v8h lo = *reinterpret_cast<const v8h*>(row + k0);
  v8h hi = *reinterpret_cast<const v8h*>(row + k0 + 16);
  v16h r;
#pragma unroll
  for (int i = 0; i < 8; ++i) { r[i] = lo[i]; r[i + 8] = hi[i]; }
  return r;
}

// ---------------------------------------------------------------------------
// Generic f16 GEMM:  out(M,Npad) = A(M,K) @ Wt(N,K)^T   (f32 accumulate)
// K % 32 == 0, M % 16 == 0, Npad % 16 == 0.  Wt zero-padded to ceil128(Npad)
// rows.  One workgroup = 8 waves = 16 x 128 output tile; whole K staged in
// LDS (A tile 16xK, B tile 128xK) via async DMA.
// ---------------------------------------------------------------------------
__global__ __launch_bounds__(256)
void k_gemm_f16(const _Float16* __restrict__ A, const _Float16* __restrict__ Wt,
                _Float16* __restrict__ out, int M, int Npad, int K) {
  extern __shared__ _Float16 smem[];
  _Float16* As = smem;               // 16  x K
  _Float16* Bs = smem + 16 * K;      // 128 x K
  const int m0 = blockIdx.x * 16;
  const int n0 = blockIdx.y * 128;
  const int lane = threadIdx.x;      // 0..31
  const int wave = threadIdx.y;      // 0..7
  const int tid  = wave * 32 + lane;

  const vi4* gA = reinterpret_cast<const vi4*>(A  + (size_t)m0 * K);
  const vi4* gB = reinterpret_cast<const vi4*>(Wt + (size_t)n0 * K);
  vi4* sA = reinterpret_cast<vi4*>(As);
  vi4* sB = reinterpret_cast<vi4*>(Bs);
  const int nA = (16 * K) / 8;       // #16B chunks
  const int nB = (128 * K) / 8;
#if defined(ATHENA_ASYNC_LDS)
  typedef __attribute__((address_space(1))) vi4 GV4;
  typedef __attribute__((address_space(3))) vi4 LV4;
  for (int i = tid; i < nA; i += 256)
    __builtin_amdgcn_global_load_async_to_lds_b128((GV4*)(gA + i), (LV4*)(sA + i), 0, 0);
  for (int i = tid; i < nB; i += 256)
    __builtin_amdgcn_global_load_async_to_lds_b128((GV4*)(gB + i), (LV4*)(sB + i), 0, 0);
  __builtin_amdgcn_s_wait_asynccnt(0);
#else
  for (int i = tid; i < nA; i += 256) sA[i] = gA[i];
  for (int i = tid; i < nB; i += 256) sB[i] = gB[i];
#endif
  __syncthreads();

  const int nc = n0 + wave * 16;     // wave-uniform -> EXEC stays all-ones
  if (nc < Npad) {
    const int r   = lane & 15;
    const int sel = lane >> 4;
    const _Float16* arow = As + r * K;
    const _Float16* brow = Bs + (wave * 16 + (lane & 15)) * K;
    v8f acc = {0.f, 0.f, 0.f, 0.f, 0.f, 0.f, 0.f, 0.f};
    for (int kb = 0; kb < K; kb += 32) {
      v16h a = ld_frag(arow, kb + sel * 8);
      v16h b = ld_frag(brow, kb + sel * 8);
      acc = __builtin_amdgcn_wmma_f32_16x16x32_f16(false, a, false, b,
                                                   (short)0, acc, false, false);
    }
    const int n = nc + (lane & 15);
#pragma unroll
    for (int v = 0; v < 8; ++v) {
      const int m = m0 + v + sel * 8;
      out[(size_t)m * Npad + n] = (_Float16)acc[v];
    }
  }
}

// Weight convert f32 -> f16, zero-padded to (Nfill, Kfill).
__global__ void k_cvtw(const float* __restrict__ w, _Float16* __restrict__ o,
                       int Nreal, int Kreal, int Nfill, int Kfill) {
  size_t i = blockIdx.x * (size_t)blockDim.x + threadIdx.x;
  size_t total = (size_t)Nfill * Kfill;
  if (i >= total) return;
  int n = (int)(i / Kfill), k = (int)(i % Kfill);
  float v = (n < Nreal && k < Kreal) ? w[(size_t)n * Kreal + k] : 0.f;
  o[i] = (_Float16)v;
}

// x (B,C,H,W) f32 -> tok (B*H*W, C) f16, token order (b,h,w)
__global__ void k_x_to_tok(const float* __restrict__ x, _Float16* __restrict__ tok) {
  size_t i = blockIdx.x * (size_t)blockDim.x + threadIdx.x;
  size_t total = (size_t)Bn * Hn * Wn * Cn;
  if (i >= total) return;
  int c = (int)(i % Cn);
  size_t t = i / Cn;
  int w = (int)(t % Wn); int h = (int)((t / Wn) % Hn); int b = (int)(t / ((size_t)Wn * Hn));
  tok[i] = (_Float16)x[(((size_t)b * Cn + c) * Hn + h) * Wn + w];
}

// (b,h,w,c) f16 -> (b,w,h,c) f16 (column-mamba token order)
__global__ void k_bhwc_to_bwhc(const _Float16* __restrict__ in, _Float16* __restrict__ out) {
  size_t i = blockIdx.x * (size_t)blockDim.x + threadIdx.x;
  size_t total = (size_t)Bn * Hn * Wn * Cn;
  if (i >= total) return;
  int c = (int)(i % Cn);
  size_t t = i / Cn;
  int h = (int)(t % Hn); int w = (int)((t / Hn) % Wn); int b = (int)(t / ((size_t)Hn * Wn));
  out[i] = in[(((size_t)b * Hn + h) * Wn + w) * Cn + c];
}

// Spectral token pack: tok (B*H*W*8, 32) = x[b, tkn*12+g, h, w] + pe[g], K-pad to 32
__global__ void k_pack_spe(const float* __restrict__ x, const float* __restrict__ pe,
                           _Float16* __restrict__ tok) {
  size_t i = blockIdx.x * (size_t)blockDim.x + threadIdx.x;
  size_t total = (size_t)Bn * Hn * Wn * TOKN * 32;
  if (i >= total) return;
  int g = (int)(i % 32);
  size_t t2 = i / 32;
  int tkn = (int)(t2 % TOKN);
  size_t sp = t2 / TOKN;
  int w = (int)(sp % Wn); int h = (int)((sp / Wn) % Hn); int b = (int)(sp / ((size_t)Wn * Hn));
  float v = 0.f;
  if (g < GDIM) {
    int c = tkn * GDIM + g;
    v = x[(((size_t)b * Cn + c) * Hn + h) * Wn + w] + pe[g];
  }
  tok[i] = (_Float16)v;
}

// Depthwise causal conv (width 4) + bias + SiLU: xz(M,2di)[:, :di] -> xi(M,diPad)
__global__ void k_conv_silu(const _Float16* __restrict__ xz, const float* __restrict__ cw,
                            const float* __restrict__ cb, _Float16* __restrict__ xi,
                            long M, int L, int di, int diPad, int twoDi) {
  size_t i = blockIdx.x * (size_t)blockDim.x + threadIdx.x;
  size_t total = (size_t)M * diPad;
  if (i >= total) return;
  int d = (int)(i % diPad);
  size_t t = i / diPad;
  if (d >= di) { xi[i] = (_Float16)0.f; return; }
  int l = (int)(t % L);
  size_t t0 = t - l;
  float acc = cb[d];
#pragma unroll
  for (int j = 0; j < 4; ++j) {
    int li = l - 3 + j;
    if (li >= 0) acc += cw[d * 4 + j] * (float)xz[(t0 + li) * twoDi + d];
  }
  xi[(size_t)t * diPad + d] = (_Float16)siluf(acc);
}

// dt = softplus(x_dbl[:, :dtr] @ dt_w^T + dt_b)   (K tiny -> VALU)
__global__ void k_dt(const _Float16* __restrict__ xdbl, const float* __restrict__ dtw,
                     const float* __restrict__ dtb, float* __restrict__ dt,
                     long M, int di, int dtr) {
  size_t i = blockIdx.x * (size_t)blockDim.x + threadIdx.x;
  size_t total = (size_t)M * di;
  if (i >= total) return;
  int d = (int)(i % di);
  size_t t = i / di;
  float s = dtb[d];
  for (int r = 0; r < dtr; ++r) s += (float)xdbl[t * 48 + r] * dtw[d * dtr + r];
  dt[i] = (s > 20.f) ? s : log1pf(__expf(s));
}

// Selective scan, fused with +x*D and SiLU(z) gating.
// 1 block per sequence, 1 thread per channel; 16 states in registers.
__global__ void k_scan(const float* __restrict__ dt, const _Float16* __restrict__ xi,
                       const _Float16* __restrict__ xdbl, const _Float16* __restrict__ xz,
                       const float* __restrict__ A_log, const float* __restrict__ Dp,
                       _Float16* __restrict__ y, int L, int di, int diPad, int dtr, int twoDi) {
  const size_t seq = blockIdx.x;
  const int d = threadIdx.x;
  if (d >= di) {                              // K-pad lanes of y
    for (int l = 0; l < L; ++l) y[(seq * L + l) * diPad + d] = (_Float16)0.f;
    return;
  }
  float Arow[16];
#pragma unroll
  for (int s = 0; s < 16; ++s) Arow[s] = -__expf(A_log[d * 16 + s]);
  const float Dd = Dp[d];
  float h[16];
#pragma unroll
  for (int s = 0; s < 16; ++s) h[s] = 0.f;
  for (int l = 0; l < L; ++l) {
    const size_t t = seq * L + l;
    const float dtv = dt[t * di + d];
    const float xv  = (float)xi[t * diPad + d];
    const float dx  = dtv * xv;
    const _Float16* xb = xdbl + t * 48;
    float yv = 0.f;
#pragma unroll
    for (int s = 0; s < 16; ++s) {
      const float Bs = (float)xb[dtr + s];
      const float Cs = (float)xb[dtr + 16 + s];
      h[s] = __expf(dtv * Arow[s]) * h[s] + dx * Bs;
      yv += h[s] * Cs;
    }
    yv += xv * Dd;
    const float zv = (float)xz[t * twoDi + di + d];
    yv *= zv * (1.0f / (1.0f + __expf(-zv)));
    y[t * diPad + d] = (_Float16)yv;
  }
}

// GroupNorm statistics. layout 0: data is (b,w,h,96) f16 (col-mamba output).
// layout 1: data is (b,h,w,8,16) f16 (spectral output, channel=tkn*12+gi).
__global__ void k_gn_stats(const _Float16* __restrict__ data, float2* __restrict__ st,
                           int layout) {
  const int bg = blockIdx.x, b = bg >> 2, g = bg & 3;
  const int CNT = 24 * Hn * Wn;
  float s = 0.f, s2 = 0.f;
  for (int i = threadIdx.x; i < CNT; i += 256) {
    const int cc = i % 24, sp = i / 24;
    const int c = g * 24 + cc;
    float v;
    if (layout == 0) {
      const int w = sp / Hn, h = sp % Hn;
      v = (float)data[(((size_t)b * Wn + w) * Hn + h) * Cn + c];
    } else {
      const int h = sp / Wn, w = sp % Wn;
      const int tkn = c / GDIM, gi = c % GDIM;
      v = (float)data[((((size_t)b * Hn + h) * Wn + w) * TOKN + tkn) * 16 + gi];
    }
    s += v; s2 += v * v;
  }
  __shared__ float rs[256], rq[256];
  rs[threadIdx.x] = s; rq[threadIdx.x] = s2;
  __syncthreads();
  for (int o = 128; o > 0; o >>= 1) {
    if (threadIdx.x < (unsigned)o) { rs[threadIdx.x] += rs[threadIdx.x + o];
                                     rq[threadIdx.x] += rq[threadIdx.x + o]; }
    __syncthreads();
  }
  if (threadIdx.x == 0) {
    const float mu = rs[0] / CNT;
    st[bg] = make_float2(mu, rq[0] / CNT - mu * mu);
  }
}

// spa = SiLU(GN(col-output))   (reads (b,w,h,c) f16, writes (b,c,h,w) f32)
__global__ void k_gn_apply_spa(const _Float16* __restrict__ xc, const float2* __restrict__ st,
                               const float* __restrict__ gw, const float* __restrict__ gb,
                               float* __restrict__ spa) {
  size_t i = blockIdx.x * (size_t)blockDim.x + threadIdx.x;
  size_t total = (size_t)Bn * Cn * Hn * Wn;
  if (i >= total) return;
  int w = (int)(i % Wn);
  int h = (int)((i / Wn) % Hn);
  int c = (int)((i / ((size_t)Wn * Hn)) % Cn);
  int b = (int)(i / ((size_t)Wn * Hn * Cn));
  float v = (float)xc[(((size_t)b * Wn + w) * Hn + h) * Cn + c];
  float2 s = st[b * 4 + c / 24];
  float xn = (v - s.x) * rsqrtf(s.y + GN_EPS) * gw[c] + gb[c];
  spa[i] = siluf(xn);
}

// Final: out = spa*softmax(fw)[0] + SiLU(GN(spe))*softmax(fw)[1]
__global__ void k_fuse(const _Float16* __restrict__ o16, const float2* __restrict__ st,
                       const float* __restrict__ gw, const float* __restrict__ gb,
                       const float* __restrict__ spa, const float* __restrict__ fw,
                       float* __restrict__ out) {
  size_t i = blockIdx.x * (size_t)blockDim.x + threadIdx.x;
  size_t total = (size_t)Bn * Cn * Hn * Wn;
  if (i >= total) return;
  int w = (int)(i % Wn);
  int h = (int)((i / Wn) % Hn);
  int c = (int)((i / ((size_t)Wn * Hn)) % Cn);
  int b = (int)(i / ((size_t)Wn * Hn * Cn));
  const int tkn = c / GDIM, gi = c % GDIM;
  float v = (float)o16[((((size_t)b * Hn + h) * Wn + w) * TOKN + tkn) * 16 + gi];
  float2 s = st[b * 4 + c / 24];
  float xn = (v - s.x) * rsqrtf(s.y + GN_EPS) * gw[c] + gb[c];
  float spe = siluf(xn);
  float f0 = fw[0], f1 = fw[1], m = fmaxf(f0, f1);
  float e0 = __expf(f0 - m), e1 = __expf(f1 - m), inv = 1.f / (e0 + e1);
  out[i] = spa[i] * (e0 * inv) + spe * (e1 * inv);
}

// ---------------------------------------------------------------------------
// Host-side helpers
// ---------------------------------------------------------------------------
static inline unsigned gblk(size_t total) { return (unsigned)((total + 255) / 256); }
static inline int ceil128(int n) { return (n + 127) / 128 * 128; }

static void run_cvtw(hipStream_t st, const float* w, _Float16* w16,
                     int Nreal, int Kreal, int Nfill, int Kfill) {
  k_cvtw<<<gblk((size_t)Nfill * Kfill), 256, 0, st>>>(w, w16, Nreal, Kreal, Nfill, Kfill);
}

static void run_gemm(hipStream_t st, const _Float16* A, const _Float16* W16,
                     _Float16* O, long M, int Npad, int K) {
  dim3 g((unsigned)(M / 16), (unsigned)((Npad + 127) / 128));
  dim3 b(32, 8);
  size_t sm = (size_t)(16 + 128) * K * sizeof(_Float16);
  k_gemm_f16<<<g, b, sm, st>>>(A, W16, O, (int)M, Npad, K);
}

// One full Mamba pass on pre-packed tokens tok(M, Kin) f16 -> out16(M, NoutPad).
// P = {in_proj_w, conv_w, conv_b, x_proj_w, dt_proj_w, dt_proj_b, A_log, D, out_proj_w}
static void run_mamba(hipStream_t st, const float* const P[9],
                      const _Float16* tok16, int Kin,
                      long M, long NSEQ, int L, int d_model, int di, int diPad, int dtr,
                      _Float16* w16, _Float16* xz16, _Float16* xi16, _Float16* xdbl16,
                      float* dtbuf, _Float16* y16, _Float16* out16, int NoutPad) {
  const int twoDi = 2 * di;                                   // 384 or 48 (mult of 16)
  // in_proj
  run_cvtw(st, P[0], w16, twoDi, d_model, ceil128(twoDi), Kin);
  run_gemm(st, tok16, w16, xz16, M, twoDi, Kin);
  // depthwise conv + SiLU
  k_conv_silu<<<gblk((size_t)M * diPad), 256, 0, st>>>(xz16, P[1], P[2], xi16,
                                                       M, L, di, diPad, twoDi);
  // x_proj  (N = dtr+32 padded to 48)
  run_cvtw(st, P[3], w16, dtr + 32, di, 128, diPad);
  run_gemm(st, xi16, w16, xdbl16, M, 48, diPad);
  // dt_proj + softplus
  k_dt<<<gblk((size_t)M * di), 256, 0, st>>>(xdbl16, P[4], P[5], dtbuf, M, di, dtr);
  // selective scan + skip(D) + SiLU(z) gate
  k_scan<<<(unsigned)NSEQ, diPad, 0, st>>>(dtbuf, xi16, xdbl16, xz16, P[6], P[7],
                                           y16, L, di, diPad, dtr, twoDi);
  // out_proj
  run_cvtw(st, P[8], w16, d_model, di, ceil128(NoutPad), diPad);
  run_gemm(st, y16, w16, out16, M, NoutPad, diPad);
}

extern "C" void kernel_launch(void* const* d_in, const int* in_sizes, int n_in,
                              void* d_out, int out_size, void* d_ws, size_t ws_size,
                              hipStream_t stream) {
  (void)in_sizes; (void)n_in; (void)out_size; (void)ws_size;
  const float* x = (const float*)d_in[0];
  const float* rowP[9]; const float* colP[9]; const float* speP[9];
  for (int i = 0; i < 9; ++i) {
    rowP[i] = (const float*)d_in[1 + i];
    colP[i] = (const float*)d_in[10 + i];
    speP[i] = (const float*)d_in[19 + i];
  }
  const float* spa_g = (const float*)d_in[28];
  const float* spa_b = (const float*)d_in[29];
  const float* spe_g = (const float*)d_in[30];
  const float* spe_b = (const float*)d_in[31];
  const float* pe    = (const float*)d_in[32];
  const float* fwts  = (const float*)d_in[33];
  float* out = (float*)d_out;

  char* ws = (char*)d_ws;
  size_t off = 0;
  auto alloc = [&](size_t bytes) -> char* {
    char* p = ws + off;
    off += (bytes + 255) / 256 * 256;
    return p;
  };

  const long M1 = (long)Bn * Hn * Wn;        // 131072 spatial tokens
  const long M2 = M1 * TOKN;                 // 1048576 spectral tokens

  // ---- persistent region ----
  float*    spa    = (float*)   alloc(sizeof(float) * (size_t)M1 * Cn);
  _Float16* w16    = (_Float16*)alloc(2 * 98304);          // max Nfill*Kfill halves
  float2*   st_spa = (float2*)  alloc(256);
  float2*   st_spe = (float2*)  alloc(256);
  const size_t arena0 = off;

  // ---- spatial-phase arena ----
  _Float16* tok16  = (_Float16*)alloc((size_t)M1 * 96  * 2);
  _Float16* xz16   = (_Float16*)alloc((size_t)M1 * 384 * 2);
  _Float16* xi16   = (_Float16*)alloc((size_t)M1 * 192 * 2);
  _Float16* xdbl16 = (_Float16*)alloc((size_t)M1 * 48  * 2);
  float*    dtbuf  = (float*)   alloc((size_t)M1 * 192 * 4);
  _Float16* y16    = (_Float16*)alloc((size_t)M1 * 192 * 2);
  _Float16* xr16   = (_Float16*)alloc((size_t)M1 * 96  * 2);
  _Float16* xc16   = (_Float16*)alloc((size_t)M1 * 96  * 2);

  // ---- spatial: row mamba then col mamba, GN+SiLU ----
  k_x_to_tok<<<gblk((size_t)M1 * Cn), 256, 0, stream>>>(x, tok16);
  run_mamba(stream, rowP, tok16, 96, M1, (long)Bn * Hn, Wn, 96, 192, 192, 6,
            w16, xz16, xi16, xdbl16, dtbuf, y16, xr16, 96);      // xr16: (b,h,w,c)
  k_bhwc_to_bwhc<<<gblk((size_t)M1 * Cn), 256, 0, stream>>>(xr16, tok16);
  run_mamba(stream, colP, tok16, 96, M1, (long)Bn * Wn, Hn, 96, 192, 192, 6,
            w16, xz16, xi16, xdbl16, dtbuf, y16, xc16, 96);      // xc16: (b,w,h,c)
  k_gn_stats<<<Bn * 4, 256, 0, stream>>>(xc16, st_spa, 0);
  k_gn_apply_spa<<<gblk((size_t)M1 * Cn), 256, 0, stream>>>(xc16, st_spa, spa_g, spa_b, spa);

  // ---- spectral-phase arena (reuses the spatial arena; stream order serializes) ----
  off = arena0;
  _Float16* tokS   = (_Float16*)alloc((size_t)M2 * 32 * 2);
  _Float16* xzS    = (_Float16*)alloc((size_t)M2 * 48 * 2);
  _Float16* xiS    = (_Float16*)alloc((size_t)M2 * 32 * 2);
  _Float16* xdblS  = (_Float16*)alloc((size_t)M2 * 48 * 2);
  float*    dtS    = (float*)   alloc((size_t)M2 * 24 * 4);
  _Float16* yS     = (_Float16*)alloc((size_t)M2 * 32 * 2);
  _Float16* oS     = (_Float16*)alloc((size_t)M2 * 16 * 2);

  k_pack_spe<<<gblk((size_t)M2 * 32), 256, 0, stream>>>(x, pe, tokS);
  run_mamba(stream, speP, tokS, 32, M2, M1, TOKN, GDIM, 24, 32, 1,
            w16, xzS, xiS, xdblS, dtS, yS, oS, 16);              // oS: (b,h,w,8,16)
  k_gn_stats<<<Bn * 4, 256, 0, stream>>>(oS, st_spe, 1);

  // ---- fuse ----
  k_fuse<<<gblk((size_t)M1 * Cn), 256, 0, stream>>>(oS, st_spe, spe_g, spe_b,
                                                    spa, fwts, out);
}
